// Cos_RootSpike_GLM_71683004170390
// MI455X (gfx1250) — matrix operand order repaired and checked by reading, not compiled, not run on hardware
//
#include <hip/hip_runtime.h>
#include <hip/hip_bf16.h>
#include <math.h>

// Problem constants (match reference)
#define SUB_NO   20
#define T_NO     200
#define BASIS_NO 19
#define T_DATA   100000
#define E_NO     800
#define I_NO     200

typedef __attribute__((ext_vector_type(16))) _Float16 v16h;
typedef __attribute__((ext_vector_type(8)))  float    v8f;

// ---------------------------------------------------------------------------
// K1: raised-cosine basis x weight rows -> out_filters [82, 200]
// rows 0-19 : e_kern_ns, 20-39: i_kern_ns, 40-59: e_kern_s, 60-79: i_kern_s,
// row 80: hist_ns_kern, row 81: hist_s_kern
// ---------------------------------------------------------------------------
__global__ void __launch_bounds__(T_NO)
filters_kernel(const float* __restrict__ Wns,   // [20,19,2]
               const float* __restrict__ Ws,    // [20,19,2]
               const float* __restrict__ hns,   // [19]
               const float* __restrict__ hs,    // [19]
               float* __restrict__ filt)        // [82,200]
{
    const int r = blockIdx.x;      // 0..81
    const int x = threadIdx.x;     // 0..199
    const float PI = 3.14159265358979323846f;
    const float raw = 5.0f * logf((float)x + 1.0f);
    float acc = 0.0f;
    #pragma unroll
    for (int b = 0; b < BASIS_NO; ++b) {
        const float phi = 0.5f * PI * (float)b;
        const float d = raw - phi;
        const float v = (d >= -PI && d <= PI) ? (0.5f * cosf(d) + 0.5f) : 0.0f;
        float w;
        if      (r < 20) w = Wns[(r      ) * 38 + b * 2 + 0];
        else if (r < 40) w = Wns[(r - 20 ) * 38 + b * 2 + 1];
        else if (r < 60) w = Ws [(r - 40 ) * 38 + b * 2 + 0];
        else if (r < 80) w = Ws [(r - 60 ) * 38 + b * 2 + 1];
        else if (r == 80) w = hns[b];
        else              w = hs[b];
        acc += w * v;
    }
    filt[r * T_NO + x] = acc;
}

// ---------------------------------------------------------------------------
// K2a: prebuild f16 B fragments of C_syn in the wave32 B(32x16) WMMA layout.
//   fragment index: [kb][tile(2)][lane(32)] -> v16h (16 halves)
//   lane L: column n = tile*16 + (L&15), g = L>>4; element i -> K = e0+8g+i+(i>=8?8:0)
//   columns n >= SUB_NO and K >= E are zero-padded.
// ---------------------------------------------------------------------------
__global__ void __launch_bounds__(32)
bfrag_kernel(const float* __restrict__ C,   // [SUB_NO, E]
             v16h* __restrict__ frag,       // [nkb][2][32]
             int E)
{
    const int lane = threadIdx.x & 31;
    const int kb   = blockIdx.x;
    const int tile = blockIdx.y;
    const int n    = tile * 16 + (lane & 15);
    const int g    = lane >> 4;
    const int e0   = kb * 32;
    v16h b;
    #pragma unroll
    for (int i = 0; i < 16; ++i) {
        const int e = e0 + 8 * g + i + ((i >= 8) ? 8 : 0);
        const float v = (n < SUB_NO && e < E) ? C[(size_t)n * E + e] : 0.0f;
        b[i] = (_Float16)v;
    }
    frag[((size_t)kb * 2 + tile) * 32 + lane] = b;
}

// ---------------------------------------------------------------------------
// K2b: syn = S @ C.T via v_wmma_f32_16x16x32_f16.
// S: [T_DATA, E] f32 (0/1 values -> exact in f16, f32 accumulation).
// One wave per 16-row time tile; B comes from the precomputed fragments.
// A fragment (16x32 f16, wave32): lane L, m=L&15, g=L>>4;
//   element i -> K = i + 8g + (i>=8 ? 8 : 0)
// C/D fragment (16x16 f32): lane L, v -> row = v + 8*(L>>4), col = L&15
// ---------------------------------------------------------------------------
__global__ void __launch_bounds__(32)
syn_gemm_kernel(const float* __restrict__ S,
                const v16h* __restrict__ bfrag,  // [nkb][2][32]
                float* __restrict__ outp,        // [T_DATA, SUB_NO]
                int E)
{
    const int t0   = blockIdx.x * 16;
    const int lane = threadIdx.x & 31;
    const int m    = lane & 15;
    const int g    = lane >> 4;
    const int ncol = lane & 15;
    const int n1   = 16 + ncol;

    const float* __restrict__ arow = S + (size_t)(t0 + m) * (size_t)E;

    v8f acc0 = {0.f,0.f,0.f,0.f,0.f,0.f,0.f,0.f};
    v8f acc1 = {0.f,0.f,0.f,0.f,0.f,0.f,0.f,0.f};

    const int nfull = E >> 5;
    // branch-free hot loop: vectorized A loads + fragment loads + 2 WMMA
    for (int kb = 0; kb < nfull; ++kb) {
        const int e0 = kb << 5;
        float ab[16];
        {
            const float4* p0 = (const float4*)(arow + e0 + 8 * g);
            const float4* p1 = (const float4*)(arow + e0 + 16 + 8 * g);
            float4 q;
            q = p0[0]; ab[0]=q.x;  ab[1]=q.y;  ab[2]=q.z;  ab[3]=q.w;
            q = p0[1]; ab[4]=q.x;  ab[5]=q.y;  ab[6]=q.z;  ab[7]=q.w;
            q = p1[0]; ab[8]=q.x;  ab[9]=q.y;  ab[10]=q.z; ab[11]=q.w;
            q = p1[1]; ab[12]=q.x; ab[13]=q.y; ab[14]=q.z; ab[15]=q.w;
        }
        v16h a;
        #pragma unroll
        for (int i = 0; i < 16; ++i) a[i] = (_Float16)ab[i];

        const v16h b0 = bfrag[((size_t)kb * 2 + 0) * 32 + lane];
        const v16h b1 = bfrag[((size_t)kb * 2 + 1) * 32 + lane];

        acc0 = __builtin_amdgcn_wmma_f32_16x16x32_f16(false, a, false, b0,
                                                      (short)0, acc0, false, false);
        acc1 = __builtin_amdgcn_wmma_f32_16x16x32_f16(false, a, false, b1,
                                                      (short)0, acc1, false, false);
    }
    // hoisted K-tail (E = 200 only): bounds-checked A, zero-padded fragments
    if (E & 31) {
        const int e0 = nfull << 5;
        v16h a;
        #pragma unroll
        for (int i = 0; i < 16; ++i) {
            const int kl = i + 8 * g + ((i >= 8) ? 8 : 0);
            const int e  = e0 + kl;
            a[i] = (_Float16)((e < E) ? arow[e] : 0.f);
        }
        const v16h b0 = bfrag[((size_t)nfull * 2 + 0) * 32 + lane];
        const v16h b1 = bfrag[((size_t)nfull * 2 + 1) * 32 + lane];
        acc0 = __builtin_amdgcn_wmma_f32_16x16x32_f16(false, a, false, b0,
                                                      (short)0, acc0, false, false);
        acc1 = __builtin_amdgcn_wmma_f32_16x16x32_f16(false, a, false, b1,
                                                      (short)0, acc1, false, false);
    }

    #pragma unroll
    for (int v = 0; v < 8; ++v) {
        const int trow = t0 + v + 8 * g;
        float* orow = outp + (size_t)trow * SUB_NO;
        orow[ncol] = acc0[v];
        if (n1 < SUB_NO) orow[n1] = acc1[v];
    }
}

// ---------------------------------------------------------------------------
// K3: per-channel causal conv (kernel length 200), LDS-tiled.
// Halo staging uses gfx1250 async copy (GLOBAL_LOAD_ASYNC_TO_LDS_B32):
// per-lane global address -> per-lane LDS offset, tracked by ASYNCcnt.
// ---------------------------------------------------------------------------
__global__ void __launch_bounds__(256)
conv_kernel(const float* __restrict__ syn_e,
            const float* __restrict__ syn_i,
            const float* __restrict__ filt,     // [82,200]
            float* __restrict__ syn_ns,
            float* __restrict__ syn_s)
{
    __shared__ float se[256 + T_NO - 1];
    __shared__ float si[256 + T_NO - 1];
    __shared__ float ke_ns[T_NO], ki_ns[T_NO], ke_s[T_NO], ki_s[T_NO];

    const int k   = blockIdx.y;
    const int t0  = blockIdx.x * 256;
    const int tid = threadIdx.x;

    for (int idx = tid; idx < 256 + T_NO - 1; idx += 256) {
        const int t = t0 - (T_NO - 1) + idx;
        if (t >= 0 && t < T_DATA) {
            // LDS byte offset = low 32 bits of the flat shared address
            // (aperture lives entirely in the high 32 bits).
            unsigned lse = (unsigned)(size_t)&se[idx];
            unsigned lsi = (unsigned)(size_t)&si[idx];
            const float* ge = syn_e + (size_t)t * SUB_NO + k;
            const float* gi = syn_i + (size_t)t * SUB_NO + k;
            asm volatile("global_load_async_to_lds_b32 %0, %1, off"
                         :: "v"(lse), "v"(ge) : "memory");
            asm volatile("global_load_async_to_lds_b32 %0, %1, off"
                         :: "v"(lsi), "v"(gi) : "memory");
        } else {
            se[idx] = 0.f;
            si[idx] = 0.f;
        }
    }
    if (tid < T_NO) {
        ke_ns[tid] = filt[(      k) * T_NO + tid];
        ki_ns[tid] = filt[(20 +  k) * T_NO + tid];
        ke_s [tid] = filt[(40 +  k) * T_NO + tid];
        ki_s [tid] = filt[(60 +  k) * T_NO + tid];
    }
    asm volatile("s_wait_asynccnt 0" ::: "memory");
    __syncthreads();

    const int t = t0 + tid;
    if (t < T_DATA) {
        float ans = 0.f, as = 0.f;
        #pragma unroll 4
        for (int j = 0; j < T_NO; ++j) {
            const float xe = se[tid + (T_NO - 1) - j];
            const float xi = si[tid + (T_NO - 1) - j];
            ans += ke_ns[j] * xe + ki_ns[j] * xi;
            as  += ke_s [j] * xe + ki_s [j] * xi;
        }
        syn_ns[(size_t)t * SUB_NO + k] = ans;
        syn_s [(size_t)t * SUB_NO + k] = as;
    }
}

// ---------------------------------------------------------------------------
// K4: history conv (1-sample delay) + leaf->root tree + outputs.
// ---------------------------------------------------------------------------
__global__ void __launch_bounds__(256)
tree_kernel(const float* __restrict__ syn_ns,
            const float* __restrict__ syn_s,
            const float* __restrict__ Z,
            const float* __restrict__ filt,     // [82,200]
            const float* __restrict__ C_den,    // [20,20]
            const float* __restrict__ W_sub_ns,
            const float* __restrict__ W_sub_s,
            const float* __restrict__ Theta_ns,
            const float* __restrict__ Theta_s,
            const float* __restrict__ V_o,      // [1]
            float* __restrict__ final_V,
            float* __restrict__ final_Z)
{
    __shared__ float zt[256 + T_NO];            // Z[t0-200 .. t0+255]
    __shared__ float hk_ns[T_NO], hk_s[T_NO];
    __shared__ float cden[SUB_NO * SUB_NO];
    __shared__ float th_ns[SUB_NO], th_s[SUB_NO], wns2[SUB_NO], ws2[SUB_NO];

    const int t0  = blockIdx.x * 256;
    const int tid = threadIdx.x;

    for (int idx = tid; idx < 256 + T_NO; idx += 256) {
        const int t = t0 - T_NO + idx;
        zt[idx] = (t >= 0 && t < T_DATA) ? Z[t] : 0.f;
    }
    if (tid < T_NO) {
        hk_ns[tid] = filt[80 * T_NO + tid];
        hk_s [tid] = filt[81 * T_NO + tid];
    }
    for (int idx = tid; idx < SUB_NO * SUB_NO; idx += 256) cden[idx] = C_den[idx];
    if (tid < SUB_NO) {
        th_ns[tid] = Theta_ns[tid];
        th_s [tid] = Theta_s[tid];
        const float a = W_sub_ns[tid]; wns2[tid] = a * a;
        const float b = W_sub_s [tid]; ws2 [tid] = b * b;
    }
    __syncthreads();

    const int t = t0 + tid;
    if (t >= T_DATA) return;

    // history conv: out[t] = sum_m kern[m] * Z[t-1-m]
    float hn = 0.f, hs = 0.f;
    #pragma unroll 4
    for (int mm = 0; mm < T_NO; ++mm) {
        const float z = zt[tid + (T_NO - 1) - mm];
        hn += hk_ns[mm] * z;
        hs += hk_s [mm] * z;
    }

    float rns[SUB_NO], rs[SUB_NO], sn[SUB_NO], ss[SUB_NO];
    const float* nrow = syn_ns + (size_t)t * SUB_NO;
    const float* srow = syn_s  + (size_t)t * SUB_NO;
    #pragma unroll
    for (int k = 0; k < SUB_NO; ++k) {
        rns[k] = nrow[k]; rs[k] = srow[k]; sn[k] = 0.f; ss[k] = 0.f;
    }

    // leaf-to-root message passing (children always have larger index)
    #pragma unroll
    for (int k = SUB_NO - 1; k >= 1; --k) {
        float aS = rs[k]  + th_s[k];
        float aN = rns[k] + th_ns[k];
        #pragma unroll
        for (int j = 0; j < SUB_NO; ++j) {
            const float c = cden[k * SUB_NO + j];
            aS += ss[j] * c * ws2[j];
            aN += sn[j] * c * wns2[j];
        }
        ss[k] = tanhf(aS);
        sn[k] = tanhf(aN);
    }

    float aS = hs + rs[0]  + th_s[0];
    float aN = hn + rns[0] + th_ns[0];
    #pragma unroll
    for (int j = 0; j < SUB_NO; ++j) {
        const float c = cden[j];        // root row of C_den
        aS += ss[j] * c * ws2[j];
        aN += sn[j] * c * wns2[j];
    }
    final_Z[t] = 1.f / (1.f + expf(-aS));
    final_V[t] = tanhf(aN) * wns2[0] + V_o[0];
}

// ---------------------------------------------------------------------------
extern "C" void kernel_launch(void* const* d_in, const int* in_sizes, int n_in,
                              void* d_out, int out_size, void* d_ws, size_t ws_size,
                              hipStream_t stream)
{
    (void)in_sizes; (void)n_in; (void)out_size; (void)ws_size;

    const float* S_e      = (const float*)d_in[0];   // [100000,800]
    const float* S_i      = (const float*)d_in[1];   // [100000,200]
    const float* Z        = (const float*)d_in[2];   // [100000]
    const float* C_den    = (const float*)d_in[3];   // [20,20]
    const float* C_syn_e  = (const float*)d_in[4];   // [20,800]
    const float* C_syn_i  = (const float*)d_in[5];   // [20,200]
    const float* W_syn_ns = (const float*)d_in[6];   // [20,19,2]
    const float* W_syn_s  = (const float*)d_in[7];   // [20,19,2]
    const float* W_sub_ns = (const float*)d_in[8];   // [20]
    const float* W_sub_s  = (const float*)d_in[9];   // [20]
    const float* V_o      = (const float*)d_in[10];  // [1]
    const float* Theta_ns = (const float*)d_in[11];  // [20]
    const float* Theta_s  = (const float*)d_in[12];  // [20]
    const float* hist_s_w = (const float*)d_in[13];  // [19]
    const float* hist_ns_w= (const float*)d_in[14];  // [19]

    float* out     = (float*)d_out;
    float* final_V = out;                  // [100000]
    float* final_Z = out + T_DATA;         // [100000]
    float* filters = out + 2 * T_DATA;     // [82,200]

    float* ws     = (float*)d_ws;
    float* syn_e  = ws;                                 // [100000,20]
    float* syn_i  = ws + (size_t)T_DATA * SUB_NO;       // [100000,20]
    float* syn_ns = ws + (size_t)2 * T_DATA * SUB_NO;   // [100000,20]
    float* syn_s  = ws + (size_t)3 * T_DATA * SUB_NO;   // [100000,20]
    // f16 B fragments (32B-aligned: offsets are multiples of 8 floats)
    const int nkb_e = (E_NO + 31) / 32;   // 25
    const int nkb_i = (I_NO + 31) / 32;   // 7
    v16h* bfrag_e = (v16h*)(ws + (size_t)4 * T_DATA * SUB_NO);
    v16h* bfrag_i = (v16h*)(ws + (size_t)4 * T_DATA * SUB_NO + (size_t)nkb_e * 2 * 32 * 8);

    // K1: filters (writes the out_filters output region, reused by K3/K4)
    filters_kernel<<<dim3(82), dim3(T_NO), 0, stream>>>(
        W_syn_ns, W_syn_s, hist_ns_w, hist_s_w, filters);

    // K2a: precompute f16 B fragments of C_syn (zero-padded, WMMA layout)
    bfrag_kernel<<<dim3(nkb_e, 2), dim3(32), 0, stream>>>(C_syn_e, bfrag_e, E_NO);
    bfrag_kernel<<<dim3(nkb_i, 2), dim3(32), 0, stream>>>(C_syn_i, bfrag_i, I_NO);

    // K2b: WMMA GEMMs, one wave per 16 time rows
    syn_gemm_kernel<<<dim3(T_DATA / 16), dim3(32), 0, stream>>>(
        S_e, bfrag_e, syn_e, E_NO);
    syn_gemm_kernel<<<dim3(T_DATA / 16), dim3(32), 0, stream>>>(
        S_i, bfrag_i, syn_i, I_NO);

    // K3: causal group convolutions (async LDS staging)
    const int ntiles = (T_DATA + 255) / 256;
    conv_kernel<<<dim3(ntiles, SUB_NO), dim3(256), 0, stream>>>(
        syn_e, syn_i, filters, syn_ns, syn_s);

    // K4: history conv + tree + outputs
    tree_kernel<<<dim3(ntiles), dim3(256), 0, stream>>>(
        syn_ns, syn_s, Z, filters, C_den,
        W_sub_ns, W_sub_s, Theta_ns, Theta_s, V_o,
        final_V, final_Z);
}
